// Decoder_7164005450187
// MI455X (gfx1250) — matrix operand order repaired and checked by reading, not compiled, not run on hardware
//
#include <hip/hip_runtime.h>
#include <hip/hip_bf16.h>

// ---------------------------------------------------------------------------
// 2-layer LSTM decoder (B=4096, H=512, 7 days) for MI455X / gfx1250.
// Heavy GEMMs (4096x2048x512) run on v_wmma_f32_16x16x32_bf16 with fp32
// accumulation; gate nonlinearities + cell update fused in the same kernel.
// Input projection (K=4) is algebraically folded into layer-0 gate bias:
//   Weff = Wih0 @ W_in  (2048x4),  beff = Wih0@b_in + bih0 + bhh0.
// Inner-loop addressing uses marching base pointers + constant immediate
// offsets so every fragment load is base+imm global_load_b128.
// ---------------------------------------------------------------------------

typedef __bf16 bf16;
typedef __attribute__((ext_vector_type(16))) __bf16 v16bf;
typedef __attribute__((ext_vector_type(8)))  __bf16 v8bf;
typedef __attribute__((ext_vector_type(8)))  float  v8f;

#define BATCH 4096
#define H     512
#define NG    2048          // 4*H gate columns
#define DAYS  7

__device__ __forceinline__ v8f wmma_bf16(v16bf a, v16bf b, v8f c) {
  // 8 args: (neg_a, A, neg_b, B, c_mod, C, reuse_a, reuse_b)
  return __builtin_amdgcn_wmma_f32_16x16x32_bf16(false, a, false, b, (short)0, c,
                                                 false, false);
}

__device__ __forceinline__ float sigm(float x) {
  return 1.f / (1.f + __expf(-x));
}
__device__ __forceinline__ float ftanh(float x) {
  float e = __expf(2.f * x);
  return (e - 1.f) / (e + 1.f);
}

// ---------------------------------------------------------------------------
// Pack a (2048 x K) fp32 weight matrix into bf16 WMMA B-fragment order:
// frag index = (ntile*KT + ktile)*32 + lane, each lane holds 16 contiguous K
// values of row n = ntile*16 + (lane&15), at k = ktile*32 + (lane>>4)*16.
// ---------------------------------------------------------------------------
__global__ void pack_weight_kernel(const float* __restrict__ W,
                                   bf16* __restrict__ out, int K) {
  int id = blockIdx.x * 256 + threadIdx.x;
  int KT = K >> 5;
  int total = (NG / 16) * KT * 32;
  if (id >= total) return;
  int lane = id & 31;
  int ktnt = id >> 5;
  int kt = ktnt % KT;
  int nt = ktnt / KT;
  int n  = nt * 16 + (lane & 15);
  int kb = kt * 32 + (lane >> 4) * 16;
  const float* src = W + (size_t)n * K + kb;
  bf16* dst = out + (size_t)id * 16;
#pragma unroll
  for (int i = 0; i < 16; ++i) dst[i] = (bf16)src[i];
}

// Fold the K=4 input projection into layer-0 constants.
__global__ void prep_small_kernel(const float* __restrict__ Wih0,
                                  const float* __restrict__ W_in,
                                  const float* __restrict__ b_in,
                                  const float* __restrict__ bih0,
                                  const float* __restrict__ bhh0,
                                  const float* __restrict__ bih1,
                                  const float* __restrict__ bhh1,
                                  float* __restrict__ Weff,
                                  float* __restrict__ beff,
                                  float* __restrict__ beff1) {
  int n = blockIdx.x * 256 + threadIdx.x;
  if (n >= NG) return;
  float w0 = 0.f, w1 = 0.f, w2 = 0.f, w3 = 0.f, bb = 0.f;
  const float* wr = Wih0 + (size_t)n * H;
  for (int h = 0; h < H; ++h) {
    float w = wr[h];
    float4 wi = ((const float4*)W_in)[h];   // W_in is (H, 4) row-major
    w0 += w * wi.x; w1 += w * wi.y; w2 += w * wi.z; w3 += w * wi.w;
    bb += w * b_in[h];
  }
  ((float4*)Weff)[n] = make_float4(w0, w1, w2, w3);
  beff[n]  = bb + bih0[n] + bhh0[n];
  beff1[n] = bih1[n] + bhh1[n];
}

__global__ void init_state_kernel(const float* __restrict__ h,
                                  const float* __restrict__ c,
                                  const float* __restrict__ start_emb,
                                  bf16* __restrict__ h0b, bf16* __restrict__ h1b,
                                  float* __restrict__ h1f,
                                  float* __restrict__ c0, float* __restrict__ c1,
                                  float* __restrict__ cur) {
  int i = blockIdx.x * 256 + threadIdx.x;
  if (i < BATCH * H) {
    float h0v = h[i];
    float h1v = h[BATCH * H + i];
    h0b[i] = (bf16)h0v;
    h1b[i] = (bf16)h1v;
    h1f[i] = h1v;
    c0[i]  = c[i];
    c1[i]  = c[BATCH * H + i];
  }
  if (i < BATCH * 4) cur[i] = start_emb[i & 3];
}

// ---------------------------------------------------------------------------
// Fused LSTM layer: gates = sum_s A_s @ W_s^T (+ cur@Weff^T) + bias, then the
// cell update. Block = 4 waves (one per gate i/f/g/o); each wave computes a
// 64x64 pre-activation tile with 16 WMMA accumulators, stages it in LDS, then
// the block applies the nonlinearity and updates c / h.
// Grid: (BATCH/64, H/64), block 128 threads.
// ---------------------------------------------------------------------------
template <int NSRC>
__global__ __launch_bounds__(128) void lstm_layer_kernel(
    const bf16* __restrict__ A0, const bf16* __restrict__ W0,
    const bf16* __restrict__ A1, const bf16* __restrict__ W1,
    const float* __restrict__ bias,   // (2048,)
    const float* __restrict__ Weff,   // (2048,4) or nullptr
    const float* __restrict__ cur,    // (4096,4) or nullptr
    float* __restrict__ c_state,      // (4096,512) in/out
    bf16*  __restrict__ h_out_b,      // (4096,512)
    float* __restrict__ h_out_f)      // (4096,512) or nullptr
{
  __shared__ float lds[4 * 64 * 64];   // 64 KB: one 64x64 fp32 tile per gate

  const int lane = threadIdx.x & 31;
  const int wv   = threadIdx.x >> 5;   // gate id 0..3
  const int m0   = blockIdx.x * 64;
  const int j0   = blockIdx.y * 64;    // hidden-column base (0..511)
  const int lh   = lane >> 4;          // lane half
  const int ll   = lane & 15;

  const v8f zero = {0.f, 0.f, 0.f, 0.f, 0.f, 0.f, 0.f, 0.f};
  v8f acc[4][4];
#pragma unroll
  for (int mt = 0; mt < 4; ++mt)
#pragma unroll
    for (int nt = 0; nt < 4; ++nt) acc[mt][nt] = zero;

  // n-tile base within the (2048/16)=128 tile space of packed weights.
  const int ntBase = wv * 32 + blockIdx.y * 4;

  // Byte strides (compile-time immediates for the loads):
  //   A: m-tile stride = 16 rows * 512 * 2B = 16384; hi-chunk = +16B; kt += 64B
  //   B: n-tile stride = KT(16)*32*16*2B = 16384;        kt += 1024B
  constexpr int A_MT   = 16 * H * 2;        // 16384
  constexpr int B_NT   = (H / 32) * 32 * 16 * 2;  // 16384
  constexpr int A_STEP = 32 * 2;            // 64
  constexpr int B_STEP = 32 * 16 * 2;       // 1024

#pragma unroll
  for (int s = 0; s < NSRC; ++s) {
    const char* ap = (const char*)((s == 0 ? A0 : A1) +
                                   (size_t)(m0 + ll) * H + lh * 16);
    const char* bp = (const char*)(s == 0 ? W0 : W1) +
                     ((size_t)ntBase * (H / 32) * 32 + lane) * 16 * 2;
    for (int kt = 0; kt < H / 32; ++kt) {
      // A fragments: lane half lh holds K = kb+16*lh..kb+16*lh+15
      // (two adjacent 16-byte chunks), matching the packed B order.
      v16bf af[4];
#pragma unroll
      for (int mt = 0; mt < 4; ++mt) {
        v8bf lo = *(const v8bf*)(ap + mt * A_MT);
        v8bf hi = *(const v8bf*)(ap + mt * A_MT + 16);
        v16bf a;
#pragma unroll
        for (int q = 0; q < 8; ++q) { a[q] = lo[q]; a[q + 8] = hi[q]; }
        af[mt] = a;
      }
      // B fragments: pre-packed, one 32-byte contiguous read per lane.
      v16bf bfr[4];
#pragma unroll
      for (int nt = 0; nt < 4; ++nt)
        bfr[nt] = *(const v16bf*)(bp + nt * B_NT);
#pragma unroll
      for (int mt = 0; mt < 4; ++mt)
#pragma unroll
        for (int nt = 0; nt < 4; ++nt)
          acc[mt][nt] = wmma_bf16(af[mt], bfr[nt], acc[mt][nt]);
      ap += A_STEP;
      bp += B_STEP;
    }
  }

  // Spill pre-activation gates to LDS (C layout: VGPR v -> M = lh*8+v, N = ll).
  float* ldsg = lds + wv * 4096;
#pragma unroll
  for (int mt = 0; mt < 4; ++mt)
#pragma unroll
    for (int nt = 0; nt < 4; ++nt)
#pragma unroll
      for (int v = 0; v < 8; ++v)
        ldsg[(mt * 16 + lh * 8 + v) * 64 + nt * 16 + ll] = acc[mt][nt][v];
  __syncthreads();

  // Fused cell update: 4096 elements over 128 threads.
  for (int e = threadIdx.x; e < 4096; e += 128) {
    int m = e >> 6, j = e & 63;
    int row = m0 + m, col = j0 + j;
    float g4[4];
#pragma unroll
    for (int q = 0; q < 4; ++q) {
      float pre = lds[q * 4096 + e];
      int n = q * H + col;
      pre += bias[n];
      if (Weff) {  // layer 0: folded input projection, rank-4 term
        float4 we = ((const float4*)Weff)[n];
        float4 cu = ((const float4*)cur)[row];
        pre += we.x * cu.x + we.y * cu.y + we.z * cu.z + we.w * cu.w;
      }
      g4[q] = pre;
    }
    float ig = sigm(g4[0]);
    float fg = sigm(g4[1]);
    float gg = ftanh(g4[2]);
    float og = sigm(g4[3]);
    size_t idx = (size_t)row * H + col;
    float cn = fg * c_state[idx] + ig * gg;
    float hn = og * ftanh(cn);
    c_state[idx] = cn;
    h_out_b[idx] = (bf16)hn;
    if (h_out_f) h_out_f[idx] = hn;
  }
}

// ---------------------------------------------------------------------------
// Output head: 9 dot products of length 512 per row (mu/log_sigma x4, p).
// One wave per row; lane-strided accumulation + xor-shuffle reduction.
// ---------------------------------------------------------------------------
__global__ __launch_bounds__(256) void head_kernel(
    const float* __restrict__ h1f,
    const float* __restrict__ W_out, const float* __restrict__ b_out,
    const float* __restrict__ W_p,   const float* __restrict__ b_p,
    float* __restrict__ out, float* __restrict__ cur, int day)
{
  int lane = threadIdx.x & 31;
  int row  = blockIdx.x * 8 + (threadIdx.x >> 5);
  float acc[9];
#pragma unroll
  for (int t = 0; t < 9; ++t) acc[t] = 0.f;
  const float* hr = h1f + (size_t)row * H;
  for (int k = lane; k < H; k += 32) {
    float hv = hr[k];
#pragma unroll
    for (int t = 0; t < 8; ++t) acc[t] += hv * W_out[t * H + k];
    acc[8] += hv * W_p[k];
  }
#pragma unroll
  for (int off = 16; off > 0; off >>= 1)
#pragma unroll
    for (int t = 0; t < 9; ++t) acc[t] += __shfl_xor(acc[t], off, 32);

  if (lane == 0) {
    float* mu_out = out;                        // (B, DAYS, 4)
    float* sg_out = out + BATCH * DAYS * 4;     // (B, DAYS, 4)
    float* p_out  = out + 2 * BATCH * DAYS * 4; // (B, DAYS, 1)
#pragma unroll
    for (int t = 0; t < 4; ++t) {
      float mu = acc[2 * t]     + b_out[2 * t];
      float ls = acc[2 * t + 1] + b_out[2 * t + 1];
      ls = fminf(2.f, fmaxf(-6.f, ls));
      mu_out[row * (DAYS * 4) + day * 4 + t] = mu;
      sg_out[row * (DAYS * 4) + day * 4 + t] = __expf(ls);
      cur[row * 4 + t] = mu;                    // autoregressive feedback
    }
    float pv = acc[8] + b_p[0];
    p_out[row * DAYS + day] = sigm(pv);
  }
}

// ---------------------------------------------------------------------------
extern "C" void kernel_launch(void* const* d_in, const int* in_sizes, int n_in,
                              void* d_out, int out_size, void* d_ws, size_t ws_size,
                              hipStream_t stream) {
  (void)in_sizes; (void)n_in; (void)out_size; (void)ws_size;
  const float* h         = (const float*)d_in[0];
  const float* c         = (const float*)d_in[1];
  const float* start_emb = (const float*)d_in[2];
  const float* W_in      = (const float*)d_in[3];
  const float* b_in      = (const float*)d_in[4];
  const float* Wih0      = (const float*)d_in[5];
  const float* Whh0      = (const float*)d_in[6];
  const float* bih0      = (const float*)d_in[7];
  const float* bhh0      = (const float*)d_in[8];
  const float* Wih1      = (const float*)d_in[9];
  const float* Whh1      = (const float*)d_in[10];
  const float* bih1      = (const float*)d_in[11];
  const float* bhh1      = (const float*)d_in[12];
  const float* W_out     = (const float*)d_in[13];
  const float* b_out     = (const float*)d_in[14];
  const float* W_p       = (const float*)d_in[15];
  const float* b_p       = (const float*)d_in[16];

  char* ws = (char*)d_ws;
  size_t off = 0;
  auto alloc = [&](size_t bytes) -> char* {
    char* p = ws + off;
    off = (off + bytes + 255) & ~(size_t)255;
    return p;
  };
  bf16* Whh0p = (bf16*)alloc((size_t)NG * H * 2);
  bf16* Wih1p = (bf16*)alloc((size_t)NG * H * 2);
  bf16* Whh1p = (bf16*)alloc((size_t)NG * H * 2);
  bf16* h0b[2] = {(bf16*)alloc((size_t)BATCH * H * 2),
                  (bf16*)alloc((size_t)BATCH * H * 2)};
  bf16* h1b[2] = {(bf16*)alloc((size_t)BATCH * H * 2),
                  (bf16*)alloc((size_t)BATCH * H * 2)};
  float* h1f  = (float*)alloc((size_t)BATCH * H * 4);
  float* c0   = (float*)alloc((size_t)BATCH * H * 4);
  float* c1   = (float*)alloc((size_t)BATCH * H * 4);
  float* Weff = (float*)alloc((size_t)NG * 4 * 4);
  float* beff = (float*)alloc((size_t)NG * 4);
  float* beff1= (float*)alloc((size_t)NG * 4);
  float* cur  = (float*)alloc((size_t)BATCH * 4 * 4);

  // One-time prep (deterministic, re-run each call).
  int packThreads = (NG / 16) * (H / 32) * 32;
  pack_weight_kernel<<<(packThreads + 255) / 256, 256, 0, stream>>>(Whh0, Whh0p, H);
  pack_weight_kernel<<<(packThreads + 255) / 256, 256, 0, stream>>>(Wih1, Wih1p, H);
  pack_weight_kernel<<<(packThreads + 255) / 256, 256, 0, stream>>>(Whh1, Whh1p, H);
  prep_small_kernel<<<NG / 256, 256, 0, stream>>>(Wih0, W_in, b_in, bih0, bhh0,
                                                  bih1, bhh1, Weff, beff, beff1);
  init_state_kernel<<<(BATCH * H + 255) / 256, 256, 0, stream>>>(
      h, c, start_emb, h0b[0], h1b[0], h1f, c0, c1, cur);

  dim3 ggrid(BATCH / 64, H / 64);  // (64, 8)
  for (int day = 0; day < DAYS; ++day) {
    int p = day & 1;
    // Layer 0: gates = h0@Whh0^T + cur@Weff^T + beff
    lstm_layer_kernel<1><<<ggrid, 128, 0, stream>>>(
        h0b[p], Whh0p, nullptr, nullptr, beff, Weff, cur, c0, h0b[1 - p], nullptr);
    // Layer 1: gates = h0_new@Wih1^T + h1@Whh1^T + beff1  (K=1024 fused)
    lstm_layer_kernel<2><<<ggrid, 128, 0, stream>>>(
        h0b[1 - p], Wih1p, h1b[p], Whh1p, beff1, nullptr, nullptr, c1,
        h1b[1 - p], h1f);
    // Output head: mu/sigma/p + autoregressive feedback into cur.
    head_kernel<<<BATCH / 8, 256, 0, stream>>>(h1f, W_out, b_out, W_p, b_p,
                                               (float*)d_out, cur, day);
  }
}
